// Encoder_19189913878759
// MI455X (gfx1250) — matrix operand order repaired and checked by reading, not compiled, not run on hardware
//
#include <hip/hip_runtime.h>

// Problem constants (from reference): B=512, T=128, N=128, H=256
#define B_TOT 512
#define T_LEN 128
#define N_IN  128
#define H_DIM 256

typedef __attribute__((ext_vector_type(16))) _Float16 v16h;
typedef __attribute__((ext_vector_type(8)))  _Float16 v8h;
typedef __attribute__((ext_vector_type(8)))  float    v8f;

// ---------------------------------------------------------------------------
// helpers
// ---------------------------------------------------------------------------
__device__ __forceinline__ float fast_tanh(float x) {
  x = fminf(15.0f, fmaxf(-15.0f, x));
  float t = __expf(2.0f * x);                 // v_exp_f32 (TRANS pipe)
  return __fdividef(t - 1.0f, t + 1.0f);      // v_rcp_f32 + mul
}

// A-fragment (16x32 f16) from row-major f16 LDS array, row stride `ld`.
// ISA layout: lanes 0-15 row M=lane: v[0..3]=K0..7, v[4..7]=K16..23;
// lanes 16-31 same rows: K8..15 / K24..31.
__device__ __forceinline__ v16h load_A(const _Float16* base, int ld, int row0, int k0) {
  int lane = threadIdx.x & 31;
  int m    = row0 + (lane & 15);
  int klo  = (lane < 16) ? 0 : 8;
  const _Float16* r = base + m * ld + k0 + klo;
  v8h lo = *(const v8h*)(r);        // 16B contiguous -> ds_load_b128
  v8h hi = *(const v8h*)(r + 16);
  v16h a;
#pragma unroll
  for (int e = 0; e < 8; ++e) { a[e] = lo[e]; a[e + 8] = hi[e]; }
  return a;
}

// B-fragment from pre-swizzled weights: fragment (kt,nt) stored as 32 lanes x
// 16 contiguous halfs -> a single 32B contiguous LDS read per lane.
__device__ __forceinline__ v16h load_Bsw(const _Float16* Wsw, int numKt, int kt, int nt) {
  int lane = threadIdx.x & 31;
  return *(const v16h*)(Wsw + (((size_t)((nt * numKt + kt) * 32 + lane)) << 4));
}

__device__ __forceinline__ v8f wmma_f16(v16h a, v16h b, v8f c) {
  // (neg_a, A, neg_b, B, c_mod, C, reuse_a, reuse_b) -> v_wmma_f32_16x16x32_f16
  return __builtin_amdgcn_wmma_f32_16x16x32_f16(false, a, false, b, (short)0, c,
                                                false, false);
}

// Convert f32 row-major weight W[Ktot][Ntot] into WMMA-B fragment order in LDS:
// element (kt,nt,lane,e) at flat ((nt*numKt+kt)*32+lane)*16+e holds
// W[kt*32 + (e&7) + ((e>>3)<<4) + (lane<16?0:8)][nt*16 + (lane&15)]  (B layout, K=32 tiles).
__device__ void swizzle_weight(const float* __restrict__ W, _Float16* Wsw,
                               int Ktot, int Ntot) {
  int numKt = Ktot >> 5;
  int total = Ktot * Ntot;
  for (int f = threadIdx.x; f < total; f += blockDim.x) {
    int e  = f & 15;
    int L  = (f >> 4) & 31;
    int r  = f >> 9;
    int kt = r % numKt;
    int nt = r / numKt;
    int k  = kt * 32 + (e & 7) + ((e >> 3) << 4) + ((L < 16) ? 0 : 8);
    int n  = nt * 16 + (L & 15);
    Wsw[f] = (_Float16)W[(size_t)k * Ntot + n];
  }
}

// ---------------------------------------------------------------------------
// Kernel 1: pre[b,i,t] = sum_s data[b,s,i] * W1[s,t] + b1[t]   (B,N,T) f32
// One block per batch; 8 waves each own one 16-row M tile.
// LDS: dataT f16 (32KB) + W1 swizzled f16 (32KB) + b1 (512B)
// ---------------------------------------------------------------------------
#define PRE_SMEM (32768 + 32768 + 512)

__global__ void __launch_bounds__(256, 1)
pre_kernel(const float* __restrict__ data, const float* __restrict__ W1,
           const float* __restrict__ b1, float* __restrict__ pre) {
  extern __shared__ char smem[];
  _Float16* dataT = (_Float16*)smem;             // [128][128] (i-major)
  _Float16* W1S   = (_Float16*)(smem + 32768);   // swizzled 128x128
  float*    b1s   = (float*)(smem + 65536);      // [128]

  int b = blockIdx.x;
  const float* db = data + (size_t)b * T_LEN * N_IN;

  // transpose data[b] (s,i) -> dataT[i][s], f32 -> f16
  for (int f = threadIdx.x; f < T_LEN * N_IN; f += blockDim.x) {
    int s = f >> 7, i = f & 127;
    dataT[i * 128 + s] = (_Float16)db[f];
  }
  swizzle_weight(W1, W1S, T_LEN, T_LEN);   // K=128 (numKt=4), N=128 (8 tiles)
  for (int f = threadIdx.x; f < T_LEN; f += blockDim.x) b1s[f] = b1[f];
  __syncthreads();

  int lane = threadIdx.x & 31;
  int w    = threadIdx.x >> 5;             // M tile = rows [16w, 16w+16)
  for (int nt = 0; nt < 8; ++nt) {
    float bias = b1s[nt * 16 + (lane & 15)];
    v8f acc;
#pragma unroll
    for (int r = 0; r < 8; ++r) acc[r] = bias;
#pragma unroll
    for (int kt = 0; kt < 4; ++kt)
      acc = wmma_f16(load_A(dataT, 128, w * 16, kt * 32),
                     load_Bsw(W1S, 4, kt, nt), acc);
#pragma unroll
    for (int r = 0; r < 8; ++r) {
      int i = w * 16 + r + ((lane < 16) ? 0 : 8);
      int t = nt * 16 + (lane & 15);
      pre[((size_t)b * N_IN + i) * T_LEN + t] = acc[r];
    }
  }
}

// ---------------------------------------------------------------------------
// Kernel 2: persistent RNN + attention scan.
// 32 blocks x 512 threads (16 waves); block owns batches [16*blk, 16*blk+16).
// All recurrence weights live in LDS (f16, WMMA-swizzled) for all 128 steps.
// LDS budget: 256KB weights + ~38KB tiles = ~294KB < 320KB/WGP.
// ---------------------------------------------------------------------------
#define OFF_WXS   0          // 128x256 f16  = 65536 B
#define OFF_WHS   65536      // 256x256 f16  = 131072 B
#define OFF_W2S   196608     // 256x128 f16  = 65536 B
#define OFF_H0    262144     // 16x256 f16   = 8192 B
#define OFF_H1    270336     // 16x256 f16   = 8192 B
#define OFF_XT    278528     // 16x128 f16   = 4096 B
#define OFF_Q     282624     // 16x128 f32   = 8192 B
#define OFF_E     290816     // 16x128 f32   = 8192 B
#define OFF_B     299008     // 256 f32      = 1024 B
#define OFF_B2    300032     // 128 f32      = 512 B
#define OFF_WV    300544     // 128 f32      = 512 B
#define MAIN_SMEM 301056

__global__ void __launch_bounds__(512, 1)
rnn_attn_kernel(const float* __restrict__ data, const float* __restrict__ h0,
                const float* __restrict__ Wx, const float* __restrict__ Wh,
                const float* __restrict__ bvec, const float* __restrict__ W2,
                const float* __restrict__ b2, const float* __restrict__ Wv,
                const float* __restrict__ bv, const float* __restrict__ pre,
                float* __restrict__ out) {
  extern __shared__ char smem[];
  _Float16* WxS  = (_Float16*)(smem + OFF_WXS);
  _Float16* WhS  = (_Float16*)(smem + OFF_WHS);
  _Float16* W2S  = (_Float16*)(smem + OFF_W2S);
  _Float16* hb0  = (_Float16*)(smem + OFF_H0);
  _Float16* hb1  = (_Float16*)(smem + OFF_H1);
  _Float16* xt   = (_Float16*)(smem + OFF_XT);
  float*    q_s  = (float*)(smem + OFF_Q);
  float*    e_s  = (float*)(smem + OFF_E);
  float*    b_s  = (float*)(smem + OFF_B);
  float*    b2_s = (float*)(smem + OFF_B2);
  float*    wv_s = (float*)(smem + OFF_WV);

  const int tid  = threadIdx.x;
  const int lane = tid & 31;
  const int w    = tid >> 5;          // wave id 0..15
  const int b0   = blockIdx.x * 16;   // first batch of this block

  // ---- prologue: stage weights/biases/h0 into LDS ----
  swizzle_weight(Wx, WxS, N_IN, H_DIM);   // numKt=4,  16 N-tiles
  swizzle_weight(Wh, WhS, H_DIM, H_DIM);  // numKt=8,  16 N-tiles
  swizzle_weight(W2, W2S, H_DIM, T_LEN);  // numKt=8,   8 N-tiles
  for (int f = tid; f < H_DIM; f += blockDim.x) b_s[f] = bvec[f];
  for (int f = tid; f < T_LEN; f += blockDim.x) { b2_s[f] = b2[f]; wv_s[f] = Wv[f]; }
  for (int f = tid; f < 16 * H_DIM; f += blockDim.x) {
    int mb = f >> 8, c = f & 255;
    hb0[f] = (_Float16)h0[(size_t)(b0 + mb) * H_DIM + c];
  }
  const float bv_val = bv[0];
  __syncthreads();

  _Float16* hIn  = hb0;
  _Float16* hOut = hb1;

  for (int t = 0; t < T_LEN; ++t) {
    // ---- load x_t tile (16 x 128) as f16 ----
#pragma unroll
    for (int j = 0; j < 4; ++j) {
      int p = tid + 512 * j;
      int mb = p >> 7, i = p & 127;
      xt[p] = (_Float16)data[((size_t)(b0 + mb) * T_LEN + t) * N_IN + i];
    }
    __syncthreads();

    // ---- GEMM1: h_new = tanh(x_t@Wx + h@Wh + b); wave w -> H cols [16w,16w+16)
    {
      float bias = b_s[w * 16 + (lane & 15)];
      v8f acc;
#pragma unroll
      for (int r = 0; r < 8; ++r) acc[r] = bias;
#pragma unroll
      for (int kt = 0; kt < 4; ++kt)                      // K over N=128
        acc = wmma_f16(load_A(xt, 128, 0, kt * 32), load_Bsw(WxS, 4, kt, w), acc);
#pragma unroll
      for (int kt = 0; kt < 8; ++kt)                      // K over H=256
        acc = wmma_f16(load_A(hIn, 256, 0, kt * 32), load_Bsw(WhS, 8, kt, w), acc);
#pragma unroll
      for (int r = 0; r < 8; ++r) {
        int mb = r + ((lane < 16) ? 0 : 8);
        hOut[mb * 256 + w * 16 + (lane & 15)] = (_Float16)fast_tanh(acc[r]);
      }
    }
    __syncthreads();

    // ---- GEMM2: q = h_new@W2 + b2; waves 0..7 -> T cols [16w,16w+16)
    if (w < 8) {
      v8f acc;
#pragma unroll
      for (int r = 0; r < 8; ++r) acc[r] = 0.0f;
#pragma unroll
      for (int kt = 0; kt < 8; ++kt)
        acc = wmma_f16(load_A(hOut, 256, 0, kt * 32), load_Bsw(W2S, 8, kt, w), acc);
      float bq = b2_s[w * 16 + (lane & 15)];
#pragma unroll
      for (int r = 0; r < 8; ++r) {
        int mb = r + ((lane < 16) ? 0 : 8);
        q_s[mb * 128 + w * 16 + (lane & 15)] = acc[r] + bq;
      }
    }
    __syncthreads();

    // ---- e[b,i] = bv + sum_t tanh(pre[b,i,t] + q[b,t]) * Wv[t]  (L2-resident pre)
#pragma unroll
    for (int j = 0; j < 4; ++j) {
      int p = tid + 512 * j;
      int mb = p >> 7, i = p & 127;
      const float4* prow = (const float4*)(pre + ((size_t)(b0 + mb) * N_IN + i) * T_LEN);
      const float*  qrow = q_s + mb * 128;
      float acc = bv_val;
#pragma unroll 4
      for (int t4 = 0; t4 < T_LEN / 4; ++t4) {
        float4 pv = prow[t4];
        int tt = t4 * 4;
        acc += fast_tanh(pv.x + qrow[tt + 0]) * wv_s[tt + 0];
        acc += fast_tanh(pv.y + qrow[tt + 1]) * wv_s[tt + 1];
        acc += fast_tanh(pv.z + qrow[tt + 2]) * wv_s[tt + 2];
        acc += fast_tanh(pv.w + qrow[tt + 3]) * wv_s[tt + 3];
      }
      e_s[p] = acc;
    }
    __syncthreads();

    // ---- softmax over i (n=128); wave w handles row mb=w (wave32 shuffles)
    {
      float v0 = e_s[w * 128 + lane];
      float v1 = e_s[w * 128 + lane + 32];
      float v2 = e_s[w * 128 + lane + 64];
      float v3 = e_s[w * 128 + lane + 96];
      float m = fmaxf(fmaxf(v0, v1), fmaxf(v2, v3));
#pragma unroll
      for (int s = 16; s > 0; s >>= 1) m = fmaxf(m, __shfl_xor(m, s, 32));
      float x0 = __expf(v0 - m), x1 = __expf(v1 - m);
      float x2 = __expf(v2 - m), x3 = __expf(v3 - m);
      float sum = x0 + x1 + x2 + x3;
#pragma unroll
      for (int s = 16; s > 0; s >>= 1) sum += __shfl_xor(sum, s, 32);
      float inv = __fdividef(1.0f, sum);
      e_s[w * 128 + lane]      = x0 * inv;   // overwrite e with alpha
      e_s[w * 128 + lane + 32] = x1 * inv;
      e_s[w * 128 + lane + 64] = x2 * inv;
      e_s[w * 128 + lane + 96] = x3 * inv;
    }
    __syncthreads();

    // ---- output (faithful to the reference's (T,B,n)->(B,T,n) reshape):
    // alpha from (t, batch bg) scales out row beta = 4t + (bg>>7), tau = bg & 127
#pragma unroll
    for (int j = 0; j < 4; ++j) {
      int p = tid + 512 * j;
      int mb = p >> 7, i = p & 127;
      int bg   = b0 + mb;
      int beta = 4 * t + (bg >> 7);
      int tau  = bg & 127;
      size_t off = ((size_t)beta * T_LEN + tau) * N_IN + i;
      out[off] = data[off] * e_s[mb * 128 + i];
    }
    __syncthreads();

    _Float16* tmp = hIn; hIn = hOut; hOut = tmp;   // ping-pong h buffers
  }
}

// ---------------------------------------------------------------------------
extern "C" void kernel_launch(void* const* d_in, const int* in_sizes, int n_in,
                              void* d_out, int out_size, void* d_ws, size_t ws_size,
                              hipStream_t stream) {
  const float* data = (const float*)d_in[0];
  const float* h0   = (const float*)d_in[1];
  const float* Wx   = (const float*)d_in[2];
  const float* Wh   = (const float*)d_in[3];
  const float* b    = (const float*)d_in[4];
  const float* W1   = (const float*)d_in[5];
  const float* b1   = (const float*)d_in[6];
  const float* W2   = (const float*)d_in[7];
  const float* b2   = (const float*)d_in[8];
  const float* Wv   = (const float*)d_in[9];
  const float* bv   = (const float*)d_in[10];
  (void)in_sizes; (void)n_in; (void)out_size; (void)ws_size;

  float* out = (float*)d_out;
  float* pre = (float*)d_ws;   // B*N*T f32 = 32 MB scratch

  // Opt in to >64KB dynamic LDS (host-side, not captured by the graph).
  (void)hipFuncSetAttribute((const void*)pre_kernel,
                            hipFuncAttributeMaxDynamicSharedMemorySize, PRE_SMEM);
  (void)hipFuncSetAttribute((const void*)rnn_attn_kernel,
                            hipFuncAttributeMaxDynamicSharedMemorySize, MAIN_SMEM);

  pre_kernel<<<B_TOT, 256, PRE_SMEM, stream>>>(data, W1, b1, pre);
  rnn_attn_kernel<<<B_TOT / 16, 512, MAIN_SMEM, stream>>>(
      data, h0, Wx, Wh, b, W2, b2, Wv, bv, pre, out);
}
